// TransformerDecoderLayer_44409961840934
// MI455X (gfx1250) — matrix operand order repaired and checked by reading, not compiled, not run on hardware
//
#include <hip/hip_runtime.h>
#include <cstdint>
#include <cstddef>

#define DEV_INLINE __device__ __forceinline__

typedef __attribute__((ext_vector_type(16))) _Float16 v16h;
typedef __attribute__((ext_vector_type(8)))  _Float16 v8h;
typedef __attribute__((ext_vector_type(8)))  float    v8f;

static constexpr int kB     = 8;
static constexpr int kNQ    = 1024;
static constexpr int kNM    = 2048;
static constexpr int kDIM   = 512;
static constexpr int kHEADS = 8;
static constexpr int kDHEAD = 64;
static constexpr int kINNER = 512;   // HEADS*DHEAD
static constexpr int kMLP   = 2048;
static constexpr int kMQ    = kB * kNQ;   // 8192 rows
static constexpr int kMK    = kB * kNM;   // 16384 rows
static constexpr float kEPS = 1e-5f;

DEV_INLINE v8f vzero8() {
  v8f v;
#pragma unroll
  for (int i = 0; i < 8; ++i) v[i] = 0.0f;
  return v;
}

DEV_INLINE v8f wmma_f16(v16h a, v16h b, v8f c) {
  // D(16x16,f32) = A(16x32,f16) * B(32x16,f16) + C
  return __builtin_amdgcn_wmma_f32_16x16x32_f16(false, a, false, b, (short)0, c, false, false);
}

// ---- CDNA5 async global->LDS copy (GLOBAL_LOAD_ASYNC_TO_LDS_B128, ASYNCcnt) ----
// VGLOBAL encoding: VDST = LDS byte address, VADDR = 64-bit global address.
DEV_INLINE void async_ld_b128(const void* gptr, void* lptr) {
  const unsigned lds = (unsigned)(uintptr_t)lptr;            // flat LDS addr: low 32 bits
  const unsigned long long ga = (unsigned long long)(uintptr_t)gptr;
  asm volatile("global_load_async_to_lds_b128 %0, %1, off" ::"v"(lds), "v"(ga) : "memory");
}
DEV_INLINE void async_wait0() {
  asm volatile("s_wait_asynccnt 0x0" ::: "memory");
}

// ---- DPP16 butterfly reduction across a 16-lane half-wave (VALU, no LDS) ----
// quad_perm(1,0,3,2)=0xB1 (xor1), quad_perm(2,3,0,1)=0x4E (xor2),
// row_half_mirror=0x141 (lane^7 in oct), row_mirror=0x140 (lane^15 in row16)
template <int CTRL>
DEV_INLINE float dppf(float x) {
  return __int_as_float(__builtin_amdgcn_mov_dpp(__float_as_int(x), CTRL, 0xf, 0xf, true));
}
DEV_INLINE float red16_max(float x) {
  x = fmaxf(x, dppf<0xB1>(x));
  x = fmaxf(x, dppf<0x4E>(x));
  x = fmaxf(x, dppf<0x141>(x));
  x = fmaxf(x, dppf<0x140>(x));
  return x;
}
DEV_INLINE float red16_sum(float x) {
  x += dppf<0xB1>(x);
  x += dppf<0x4E>(x);
  x += dppf<0x141>(x);
  x += dppf<0x140>(x);
  return x;
}

// Build v16h from 16 contiguous halves (two 128-bit LDS loads)
DEV_INLINE v16h ldfrag16(const _Float16* p) {
  const v8h lo = *(const v8h*)p;
  const v8h hi = *(const v8h*)(p + 8);
  return __builtin_shufflevector(lo, hi, 0, 1, 2, 3, 4, 5, 6, 7, 8, 9, 10, 11, 12, 13, 14, 15);
}

// A-fragment from row-major LDS tile [16][ld] (rows m, cols k in 0..31).
// A order used here (B matches): half h -> k = 16g + h: contiguous,
// so two b128 loads at k=16g and k=16g+8.
DEV_INLINE v16h frag_a(const _Float16* s, int ld, int lane) {
  const int m = lane & 15, g = lane >> 4;
  const _Float16* p = s + m * ld + 16 * g;
  const v8h lo = *(const v8h*)p;
  const v8h hi = *(const v8h*)(p + 8);
  return __builtin_shufflevector(lo, hi, 0, 1, 2, 3, 4, 5, 6, 7, 8, 9, 10, 11, 12, 13, 14, 15);
}

// B-fragment from TRANSPOSED LDS tile [n][k] (ld = k-stride):
// half h -> B[k=16g+h][n], i.e. 16 contiguous k at row n.
DEV_INLINE v16h frag_bt(const _Float16* s, int ld, int lane) {
  const int n = lane & 15, g = lane >> 4;
  return ldfrag16(s + n * ld + 16 * g);
}

// ---------------------------------------------------------------------------
// Tiled WMMA GEMM: C[M,N] = A[M,K](f16) * B[K,N](f16) (+bias) (ReLU) -> f32 or f16
// Block: 128 threads (4 waves). Tile 64x64, BK=32. Wave w -> 32x32 subtile (2x2 WMMA).
// A tile copied with async global->LDS; B tile transposed through registers.
// ---------------------------------------------------------------------------
template <bool BIAS, bool RELU, bool OUTF16>
__global__ __launch_bounds__(128) void gemm64(
    const _Float16* __restrict__ A, const _Float16* __restrict__ Bm,
    const float* __restrict__ bias, float* __restrict__ Cf,
    _Float16* __restrict__ Ch, int M, int N, int K) {
  __shared__ _Float16 sA[64][40];   // [m][k]  (32 + 8 pad)
  __shared__ _Float16 sBt[64][40];  // [n][k]  transposed (32 + 8 pad)

  const int tid  = threadIdx.x;
  const int lane = tid & 31, wave = tid >> 5;
  const int wr = wave >> 1, wc = wave & 1;
  const int rowBase = blockIdx.y * 64, colBase = blockIdx.x * 64;

  v8f acc[2][2];
#pragma unroll
  for (int i = 0; i < 2; ++i)
#pragma unroll
    for (int j = 0; j < 2; ++j) acc[i][j] = vzero8();

  const int ar = tid >> 1, ac = (tid & 1) * 16;   // A: 64 rows, 2 threads/row
  const int br = tid >> 2, bc = (tid & 3) * 16;   // B: 32 k-rows, 4 threads/row

  for (int kk = 0; kk < K; kk += 32) {
    // A tile: async DMA straight into LDS (no VGPR staging)
    const _Float16* ag = A + (size_t)(rowBase + ar) * K + kk + ac;
    async_ld_b128(ag,     &sA[ar][ac]);
    async_ld_b128(ag + 8, &sA[ar][ac + 8]);
    // B tile: load, transpose into LDS
    const _Float16* bg = Bm + (size_t)(kk + br) * N + colBase + bc;
    const v8h b0 = *(const v8h*)(bg);
    const v8h b1 = *(const v8h*)(bg + 8);
#pragma unroll
    for (int e = 0; e < 8; ++e) {
      sBt[bc + e][br]     = b0[e];
      sBt[bc + 8 + e][br] = b1[e];
    }
    if (kk + 32 < K) {  // prefetch next K tiles (global_prefetch_b8)
      __builtin_prefetch(ag + 32, 0, 1);
      __builtin_prefetch(bg + (size_t)32 * N, 0, 1);
    }
    async_wait0();
    __syncthreads();

    v16h af[2], bf[2];
#pragma unroll
    for (int i = 0; i < 2; ++i) af[i] = frag_a(&sA[wr * 32 + 16 * i][0], 40, lane);
#pragma unroll
    for (int j = 0; j < 2; ++j) bf[j] = frag_bt(&sBt[wc * 32 + 16 * j][0], 40, lane);
#pragma unroll
    for (int i = 0; i < 2; ++i)
#pragma unroll
      for (int j = 0; j < 2; ++j) acc[i][j] = wmma_f16(af[i], bf[j], acc[i][j]);
    __syncthreads();
  }

  // C/D layout: VGPR r, lanes 0-15 -> row r, lanes 16-31 -> row r+8; col = lane&15
  const int g = lane >> 4, cn = lane & 15;
#pragma unroll
  for (int i = 0; i < 2; ++i)
#pragma unroll
    for (int j = 0; j < 2; ++j) {
      const int C = colBase + wc * 32 + 16 * j + cn;
      float bv = 0.0f;
      if (BIAS) bv = bias[C];
#pragma unroll
      for (int r = 0; r < 8; ++r) {
        const int R = rowBase + wr * 32 + 16 * i + r + 8 * g;
        float v = acc[i][j][r];
        if (BIAS) v += bv;
        if (RELU) v = fmaxf(v, 0.0f);
        if (OUTF16) Ch[(size_t)R * N + C] = (_Float16)v;
        else        Cf[(size_t)R * N + C] = v;
      }
    }
}

// ---------------------------------------------------------------------------
// Flash attention: one wave handles a 16-query tile for one (batch, head).
// K/V tiles arrive via async global->LDS DMA; S = Q*K^T (4 WMMA), DPP softmax,
// O += P*V (4 WMMA). Q/K/V/O are [rows, INNER] f16. Scale folded into Q (1/8).
// ---------------------------------------------------------------------------
__global__ __launch_bounds__(32) void attn32(
    const _Float16* __restrict__ Q, const _Float16* __restrict__ K,
    const _Float16* __restrict__ V, _Float16* __restrict__ O,
    int NQtot, int NKtot) {
  __shared__ _Float16 sQ[16][72];
  __shared__ _Float16 sK[32][72];
  __shared__ _Float16 sV[32][72];
  __shared__ _Float16 sP[16][40];

  const int lane = threadIdx.x;
  const int qt = blockIdx.x, head = blockIdx.y, b = blockIdx.z;
  const int qBase = qt * 16;

  // Load 16x64 Q tile via async DMA (each lane: half a row = 32 halves)
  {
    const int r = lane >> 1, c0 = (lane & 1) * 32;
    const _Float16* qg = Q + ((size_t)(b * NQtot + qBase + r)) * kINNER + head * kDHEAD + c0;
#pragma unroll
    for (int u = 0; u < 4; ++u) async_ld_b128(qg + 8 * u, &sQ[r][c0 + 8 * u]);
  }
  async_wait0();
  __syncthreads();
  v16h qa[2];
  qa[0] = frag_a(&sQ[0][0], 72, lane) * (_Float16)0.125f;   // fold 1/sqrt(64) into Q
  qa[1] = frag_a(&sQ[0][32], 72, lane) * (_Float16)0.125f;

  v8f oacc[4];
#pragma unroll
  for (int j = 0; j < 4; ++j) oacc[j] = vzero8();
  float rm[8], rl[8];
#pragma unroll
  for (int r = 0; r < 8; ++r) { rm[r] = -3.0e38f; rl[r] = 0.0f; }

  const int nkt = NKtot / 32;
  for (int kt = 0; kt < nkt; ++kt) {
    const size_t krow0 = (size_t)(b * NKtot + kt * 32);
    const _Float16* kg = K + (krow0 + lane) * kINNER + head * kDHEAD;
    const _Float16* vg = V + (krow0 + lane) * kINNER + head * kDHEAD;
#pragma unroll
    for (int u = 0; u < 8; ++u) {
      async_ld_b128(kg + 8 * u, &sK[lane][8 * u]);
      async_ld_b128(vg + 8 * u, &sV[lane][8 * u]);
    }
    async_wait0();
    __syncthreads();

    // S(16x32) = Q*K^T.  B-fragment of K^T is K[key][dh-slice], contiguous in sK.
    v8f s0 = vzero8(), s1 = vzero8();
    {
      const int n = lane & 15, g16 = (lane >> 4) * 16;
#pragma unroll
      for (int s = 0; s < 2; ++s) {
        const v16h kb0 = ldfrag16(&sK[n][s * 32 + g16]);
        const v16h kb1 = ldfrag16(&sK[16 + n][s * 32 + g16]);
        s0 = wmma_f16(qa[s], kb0, s0);
        s1 = wmma_f16(qa[s], kb1, s1);
      }
    }

    // online softmax: row statistics via DPP16 butterfly (rows live in half-waves)
    float p0[8], p1[8], sc[8];
#pragma unroll
    for (int r = 0; r < 8; ++r) {
      const float mx = red16_max(fmaxf(s0[r], s1[r]));
      const float mn = fmaxf(rm[r], mx);
      sc[r] = __expf(rm[r] - mn);
      rm[r] = mn;
      p0[r] = __expf(s0[r] - mn);
      p1[r] = __expf(s1[r] - mn);
      const float rs = red16_sum(p0[r] + p1[r]);
      rl[r] = rl[r] * sc[r] + rs;
    }
#pragma unroll
    for (int j = 0; j < 4; ++j)
#pragma unroll
      for (int r = 0; r < 8; ++r) oacc[j][r] *= sc[r];

    // stage P (16x32) as f16 for the A fragment
    {
      const int g = lane >> 4, cn = lane & 15;
#pragma unroll
      for (int r = 0; r < 8; ++r) {
        sP[r + 8 * g][cn]      = (_Float16)p0[r];
        sP[r + 8 * g][16 + cn] = (_Float16)p1[r];
      }
    }
    __syncthreads();
    const v16h pa = frag_a(&sP[0][0], 40, lane);
#pragma unroll
    for (int j = 0; j < 4; ++j) {
      v16h vb;
      {
        const int n = lane & 15, g = lane >> 4;
#pragma unroll
        for (int h = 0; h < 16; ++h) vb[h] = sV[16 * g + h][16 * j + n];
      }
      oacc[j] = wmma_f16(pa, vb, oacc[j]);
    }
    __syncthreads();  // protect sK/sV/sP before next iteration
  }

  // epilogue: O /= rowsum
  {
    const int g = lane >> 4, cn = lane & 15;
#pragma unroll
    for (int r = 0; r < 8; ++r) {
      const float inv = 1.0f / rl[r];
      const int qrow = qBase + r + 8 * g;
      _Float16* og = O + ((size_t)(b * NQtot + qrow)) * kINNER + head * kDHEAD;
#pragma unroll
      for (int j = 0; j < 4; ++j) og[16 * j + cn] = (_Float16)(oacc[j][r] * inv);
    }
  }
}

// ---------------------------------------------------------------------------
// LayerNorm(Y)*g + b + R  (per row of DIM=512), f32 out + optional f16 copy
// ---------------------------------------------------------------------------
__global__ __launch_bounds__(256) void ln_res(
    const float* __restrict__ Y, const float* __restrict__ R,
    const float* __restrict__ gam, const float* __restrict__ bet,
    float* __restrict__ Xout, _Float16* __restrict__ Hout) {
  __shared__ float sred[256];
  const int row = blockIdx.x, tid = threadIdx.x;
  const float* y = Y + (size_t)row * kDIM;
  const float v0 = y[tid], v1 = y[tid + 256];

  sred[tid] = v0 + v1;
  __syncthreads();
  for (int s = 128; s > 0; s >>= 1) {
    if (tid < s) sred[tid] += sred[tid + s];
    __syncthreads();
  }
  const float mu = sred[0] * (1.0f / kDIM);
  __syncthreads();

  const float d0 = v0 - mu, d1 = v1 - mu;
  sred[tid] = d0 * d0 + d1 * d1;
  __syncthreads();
  for (int s = 128; s > 0; s >>= 1) {
    if (tid < s) sred[tid] += sred[tid + s];
    __syncthreads();
  }
  const float var = sred[0] * (1.0f / kDIM);
  const float rstd = rsqrtf(var + kEPS);

  const float* rr = R + (size_t)row * kDIM;
  const float o0 = d0 * rstd * gam[tid] + bet[tid] + rr[tid];
  const float o1 = d1 * rstd * gam[tid + 256] + bet[tid + 256] + rr[tid + 256];
  float* xo = Xout + (size_t)row * kDIM;
  xo[tid] = o0;
  xo[tid + 256] = o1;
  if (Hout) {
    _Float16* ho = Hout + (size_t)row * kDIM;
    ho[tid] = (_Float16)o0;
    ho[tid + 256] = (_Float16)o1;
  }
}

// out = a (+ b), to f32 and/or f16
__global__ __launch_bounds__(256) void add_cvt(
    const float* __restrict__ a, const float* __restrict__ b,
    float* __restrict__ fo, _Float16* __restrict__ ho, int n) {
  const int i = blockIdx.x * 256 + threadIdx.x;
  if (i >= n) return;
  float v = a[i];
  if (b) v += b[i];
  if (fo) fo[i] = v;
  if (ho) ho[i] = (_Float16)v;
}

// ---------------------------------------------------------------------------
extern "C" void kernel_launch(void* const* d_in, const int* in_sizes, int n_in,
                              void* d_out, int out_size, void* d_ws, size_t ws_size,
                              hipStream_t stream) {
  (void)in_sizes; (void)n_in; (void)out_size; (void)ws_size;

  const float* tgt = (const float*)d_in[0];
  const float* mem = (const float*)d_in[1];
  const float* pos = (const float*)d_in[2];
  const float* qp  = (const float*)d_in[3];
  // a0: 4..10, f0: 11..16, a1: 17..23, f1: 24..29
  const float* a0_bo = (const float*)d_in[8];
  const float* a0_g  = (const float*)d_in[9];
  const float* a0_b  = (const float*)d_in[10];
  const float* f0_b1 = (const float*)d_in[12];
  const float* f0_b2 = (const float*)d_in[14];
  const float* f0_g  = (const float*)d_in[15];
  const float* f0_b  = (const float*)d_in[16];
  const float* a1_bo = (const float*)d_in[21];
  const float* a1_g  = (const float*)d_in[22];
  const float* a1_b  = (const float*)d_in[23];
  const float* f1_b1 = (const float*)d_in[25];
  const float* f1_b2 = (const float*)d_in[27];
  const float* f1_g  = (const float*)d_in[28];
  const float* f1_b  = (const float*)d_in[29];

  // ---- workspace arena ----
  char* ws = (char*)d_ws;
  size_t off = 0;
  auto take = [&](size_t bytes) -> void* {
    void* p = ws + off;
    off += (bytes + 255) & ~(size_t)255;
    return p;
  };
  _Float16* h_q0   = (_Float16*)take((size_t)kMQ * kDIM * 2);     // q-side f16 input
  _Float16* h_tgt  = (_Float16*)take((size_t)kMQ * kDIM * 2);
  _Float16* h_mem  = (_Float16*)take((size_t)kMK * kDIM * 2);
  _Float16* h_memp = (_Float16*)take((size_t)kMK * kDIM * 2);
  _Float16* Qb     = (_Float16*)take((size_t)kMQ * kINNER * 2);
  _Float16* Kb     = (_Float16*)take((size_t)kMK * kINNER * 2);
  _Float16* Vb     = (_Float16*)take((size_t)kMK * kINNER * 2);
  _Float16* Hmlp   = Kb;  // alias: FFN hidden [8192,2048] f16 over Kb+Vb (disjoint live ranges)
  _Float16* Aout   = (_Float16*)take((size_t)kMQ * kINNER * 2);
  _Float16* h_x    = (_Float16*)take((size_t)kMQ * kDIM * 2);
  float*    XA     = (float*)take((size_t)kMQ * kDIM * 4);
  float*    XB     = (float*)take((size_t)kMQ * kDIM * 4);
  float*    Ypre   = (float*)take((size_t)kMQ * kDIM * 4);
  _Float16* w_a0q  = (_Float16*)take((size_t)512 * 512 * 2);
  _Float16* w_a0k  = (_Float16*)take((size_t)512 * 512 * 2);
  _Float16* w_a0v  = (_Float16*)take((size_t)512 * 512 * 2);
  _Float16* w_a0o  = (_Float16*)take((size_t)512 * 512 * 2);
  _Float16* w_f0w1 = (_Float16*)take((size_t)512 * 2048 * 2);
  _Float16* w_f0w2 = (_Float16*)take((size_t)2048 * 512 * 2);
  _Float16* w_a1q  = (_Float16*)take((size_t)512 * 512 * 2);
  _Float16* w_a1k  = (_Float16*)take((size_t)512 * 512 * 2);
  _Float16* w_a1v  = (_Float16*)take((size_t)512 * 512 * 2);
  _Float16* w_a1o  = (_Float16*)take((size_t)512 * 512 * 2);
  _Float16* w_f1w1 = (_Float16*)take((size_t)512 * 2048 * 2);
  _Float16* w_f1w2 = (_Float16*)take((size_t)2048 * 512 * 2);

  const dim3 blk256(256), gblk(128), ablk(32);
  const dim3 agrid(kNQ / 16, kHEADS, kB);
  auto cvtw = [&](int idx, _Float16* dst, int n) {
    add_cvt<<<n / 256, blk256, 0, stream>>>((const float*)d_in[idx], nullptr, nullptr, dst, n);
  };

  // ---- input prep ----
  add_cvt<<<(kMQ * kDIM) / 256, blk256, 0, stream>>>(tgt, qp, XA, h_q0, kMQ * kDIM);   // q0 = tgt+qp
  add_cvt<<<(kMQ * kDIM) / 256, blk256, 0, stream>>>(tgt, nullptr, nullptr, h_tgt, kMQ * kDIM);
  add_cvt<<<(kMK * kDIM) / 256, blk256, 0, stream>>>(mem, pos, nullptr, h_memp, kMK * kDIM);
  add_cvt<<<(kMK * kDIM) / 256, blk256, 0, stream>>>(mem, nullptr, nullptr, h_mem, kMK * kDIM);
  cvtw(4, w_a0q, 512 * 512);  cvtw(5, w_a0k, 512 * 512);
  cvtw(6, w_a0v, 512 * 512);  cvtw(7, w_a0o, 512 * 512);
  cvtw(11, w_f0w1, 512 * 2048); cvtw(13, w_f0w2, 2048 * 512);
  cvtw(17, w_a1q, 512 * 512); cvtw(18, w_a1k, 512 * 512);
  cvtw(19, w_a1v, 512 * 512); cvtw(20, w_a1o, 512 * 512);
  cvtw(24, w_f1w1, 512 * 2048); cvtw(26, w_f1w2, 2048 * 512);

  // ---- block 0: self-attention (q=k=q0, v=tgt), residual = q0 (XA) ----
  gemm64<false, false, true><<<dim3(kINNER / 64, kMQ / 64), gblk, 0, stream>>>(
      h_q0, w_a0q, nullptr, nullptr, Qb, kMQ, kINNER, kDIM);
  gemm64<false, false, true><<<dim3(kINNER / 64, kMQ / 64), gblk, 0, stream>>>(
      h_q0, w_a0k, nullptr, nullptr, Kb, kMQ, kINNER, kDIM);
  gemm64<false, false, true><<<dim3(kINNER / 64, kMQ / 64), gblk, 0, stream>>>(
      h_tgt, w_a0v, nullptr, nullptr, Vb, kMQ, kINNER, kDIM);
  attn32<<<agrid, ablk, 0, stream>>>(Qb, Kb, Vb, Aout, kNQ, kNQ);
  gemm64<true, false, false><<<dim3(kDIM / 64, kMQ / 64), gblk, 0, stream>>>(
      Aout, w_a0o, a0_bo, Ypre, nullptr, kMQ, kDIM, kINNER);
  ln_res<<<kMQ, blk256, 0, stream>>>(Ypre, XA, a0_g, a0_b, XB, h_x);   // x1 -> XB

  // ---- block 1: FFN0, residual = x1 (XB) ----
  gemm64<true, true, true><<<dim3(kMLP / 64, kMQ / 64), gblk, 0, stream>>>(
      h_x, w_f0w1, f0_b1, nullptr, Hmlp, kMQ, kMLP, kDIM);
  gemm64<true, false, false><<<dim3(kDIM / 64, kMQ / 64), gblk, 0, stream>>>(
      Hmlp, w_f0w2, f0_b2, Ypre, nullptr, kMQ, kDIM, kMLP);
  ln_res<<<kMQ, blk256, 0, stream>>>(Ypre, XB, f0_g, f0_b, XA, h_x);   // x2 -> XA

  // ---- block 2: cross-attention (q=x2+qp, k=mem+pos, v=mem), residual = q (XB) ----
  add_cvt<<<(kMQ * kDIM) / 256, blk256, 0, stream>>>(XA, qp, XB, h_q0, kMQ * kDIM);
  gemm64<false, false, true><<<dim3(kINNER / 64, kMQ / 64), gblk, 0, stream>>>(
      h_q0, w_a1q, nullptr, nullptr, Qb, kMQ, kINNER, kDIM);
  gemm64<false, false, true><<<dim3(kINNER / 64, kMK / 64), gblk, 0, stream>>>(
      h_memp, w_a1k, nullptr, nullptr, Kb, kMK, kINNER, kDIM);
  gemm64<false, false, true><<<dim3(kINNER / 64, kMK / 64), gblk, 0, stream>>>(
      h_mem, w_a1v, nullptr, nullptr, Vb, kMK, kINNER, kDIM);
  attn32<<<agrid, ablk, 0, stream>>>(Qb, Kb, Vb, Aout, kNQ, kNM);
  gemm64<true, false, false><<<dim3(kDIM / 64, kMQ / 64), gblk, 0, stream>>>(
      Aout, w_a1o, a1_bo, Ypre, nullptr, kMQ, kDIM, kINNER);
  ln_res<<<kMQ, blk256, 0, stream>>>(Ypre, XB, a1_g, a1_b, XA, h_x);   // x3 -> XA

  // ---- block 3: FFN1, residual = x3 (XA), final output f32 -> d_out ----
  gemm64<true, true, true><<<dim3(kMLP / 64, kMQ / 64), gblk, 0, stream>>>(
      h_x, w_f1w1, f1_b1, nullptr, Hmlp, kMQ, kMLP, kDIM);
  gemm64<true, false, false><<<dim3(kDIM / 64, kMQ / 64), gblk, 0, stream>>>(
      Hmlp, w_f1w2, f1_b2, Ypre, nullptr, kMQ, kDIM, kMLP);
  ln_res<<<kMQ, blk256, 0, stream>>>(Ypre, XA, f1_g, f1_b, (float*)d_out, nullptr);
}
